// MultiHeadAttention_91087666414084
// MI455X (gfx1250) — matrix operand order repaired and checked by reading, not compiled, not run on hardware
//
#include <hip/hip_runtime.h>
#include <hip/hip_bf16.h>

// MI455X / gfx1250 multi-head attention forward.
// bf16 WMMA (v_wmma_f32_16x16x32_bf16) everywhere, fp32 accumulate.
//   1) convert x -> bf16, transpose+convert weights
//   2) fused QKV projection: weight k-tiles staged in LDS via
//      global_load_async_to_lds_b128, 80B-padded rows (conflict-free ds_load)
//   3) flash attention (online softmax, LDS relayout of P, V-frags prefetched)
//   4) output projection: weight strip staged in LDS, +bias, fp32 out
//
// Workspace (66 MB):
//   [0,16MB)   xbf  [16384,512] bf16  (reused as 'combined' after attention)
//   [16,32MB)  qbf  [B,H,T,D]   bf16  (pre-scaled 1/sqrt(D), +bq)
//   [32,48MB)  kbf  [B,H,T,D]   bf16  (+bk)
//   [48,64MB)  vtb  [B,H,D,T]   bf16  (+bv, transposed)
//   [64MB..)   wqt/wkt/wvt [H,64,512], wot [512,512] bf16 (transposed)

typedef __attribute__((ext_vector_type(16))) __bf16 v16bf;
typedef __attribute__((ext_vector_type(8)))  float  v8f;
typedef int v4i __attribute__((vector_size(16)));

#define ZV8 ((v8f){0.f,0.f,0.f,0.f,0.f,0.f,0.f,0.f})

#if defined(__has_builtin)
#  if __has_builtin(__builtin_amdgcn_global_load_async_to_lds_b128)
#    define HAVE_ASYNC_LDS 1
#  endif
#endif
#ifndef HAVE_ASYNC_LDS
#  define HAVE_ASYNC_LDS 0
#endif

#if HAVE_ASYNC_LDS
__device__ __forceinline__ void async_g2l_b128(const __bf16* g, __bf16* l) {
  __builtin_amdgcn_global_load_async_to_lds_b128(
      (__attribute__((address_space(1))) v4i*)g,
      (__attribute__((address_space(3))) v4i*)l, 0, 0);
}
__device__ __forceinline__ void wait_async() {
#if __has_builtin(__builtin_amdgcn_s_wait_asynccnt)
  __builtin_amdgcn_s_wait_asynccnt(0);
#else
  asm volatile("s_wait_asynccnt 0x0" ::: "memory");
#endif
}
#endif

__device__ __forceinline__ void stage16(const __bf16* g, __bf16* l) {
#if HAVE_ASYNC_LDS
  async_g2l_b128(g, l);
#else
  *(float4*)l = *(const float4*)g;
#endif
}

// ---- WMMA fragment helpers (ISA 7.12.2 layouts, wave32) -------------------

// A 16x32 bf16: lane = row (lane&15), half = lane>>4,
// VGPR i -> K = 16*(i>>2) + 8*half + 2*(i&3) (+1).
__device__ __forceinline__ v16bf load_a_frag(const __bf16* p, int ld) {
  union { v16bf v; unsigned u[8]; } f;
  const int lane = threadIdx.x & 31;
  const __bf16* rp = p + (lane & 15) * ld + 8 * (lane >> 4);
#pragma unroll
  for (int i = 0; i < 8; ++i) {
    const int k = 16 * (i >> 2) + 2 * (i & 3);
    f.u[i] = *(const unsigned*)(rp + k);
  }
  return f.v;
}

// B 32x16 bf16 from COLUMN-MAJOR storage (col n contiguous over K):
// lane = col (lane&15), half = lane>>4, VGPR i -> K = 16*(i>>2) + 8*half + 2*(i&3) (+1).
__device__ __forceinline__ v16bf load_b_frag(const __bf16* p, int ld) {
  union { v16bf v; unsigned u[8]; } f;
  const int lane = threadIdx.x & 31;
  const __bf16* rp = p + (lane & 15) * ld + 8 * (lane >> 4);
#pragma unroll
  for (int i = 0; i < 8; ++i)
    f.u[i] = *(const unsigned*)(rp + 16 * (i >> 2) + 2 * (i & 3));
  return f.v;
}

__device__ __forceinline__ v8f wmma_bf16(v16bf a, v16bf b, v8f c) {
  return __builtin_amdgcn_wmma_f32_16x16x32_bf16(false, a, false, b,
                                                 (short)0, c, false, false);
}

__device__ __forceinline__ float xor_max16(float v) {
#pragma unroll
  for (int m = 1; m < 16; m <<= 1) v = fmaxf(v, __shfl_xor(v, m, 32));
  return v;
}
__device__ __forceinline__ float xor_sum16(float v) {
#pragma unroll
  for (int m = 1; m < 16; m <<= 1) v += __shfl_xor(v, m, 32);
  return v;
}

// ---- Stage 0: conversions -------------------------------------------------

__global__ void mha_cvt_x(const float* __restrict__ x, __bf16* __restrict__ o) {
  const int i = (blockIdx.x * 256 + threadIdx.x) * 4;   // 16384*512 elements
  const float4 v = *(const float4*)(x + i);
  o[i + 0] = (__bf16)v.x; o[i + 1] = (__bf16)v.y;
  o[i + 2] = (__bf16)v.z; o[i + 3] = (__bf16)v.w;
}

// [H,C,D] fp32 -> [H,D,C] bf16
__global__ void mha_cvt_wqkv(const float* __restrict__ w, __bf16* __restrict__ wt) {
  const int i = blockIdx.x * 256 + threadIdx.x;         // 8*64*512
  const int h = i >> 15, rem = i & 32767;
  const int d = rem >> 9, c = rem & 511;
  wt[i] = (__bf16)w[(h * 512 + c) * 64 + d];
}

// [K=512,N=512] fp32 -> [N,K] bf16
__global__ void mha_cvt_wo(const float* __restrict__ wo, __bf16* __restrict__ wot) {
  const int i = blockIdx.x * 256 + threadIdx.x;         // 512*512
  const int n = i >> 9, k = i & 511;
  wot[i] = (__bf16)wo[k * 512 + n];
}

// ---- Stage 1: fused QKV projection ---------------------------------------
// Block = 8 waves, all on the same head h; per k-step the 3 weight tiles
// (64 cols x 32 K) staged once into LDS with 40-element (80 B) padded rows:
// lanes 0..15 hit 16 distinct banks (20*n mod 64).
#define WROW 40
#define WTILE (64 * WROW)   // 2560 elements = 5120 B per weight

__global__ __launch_bounds__(256, 1)
void mha_qkv(const __bf16* __restrict__ xbf,
             const __bf16* __restrict__ wqt, const __bf16* __restrict__ wkt,
             const __bf16* __restrict__ wvt,
             const float* __restrict__ bq, const float* __restrict__ bk,
             const float* __restrict__ bv,
             __bf16* __restrict__ qbf, __bf16* __restrict__ kbf,
             __bf16* __restrict__ vtb) {
  __shared__ __bf16 wtile[3 * WTILE];                  // 15360 B
  const int tid = threadIdx.x;
  const int wave = tid >> 5;
  const int task = blockIdx.x * 8 + wave;              // 8192 = 8 heads * 1024
  const int h = task >> 10;
  const int m0 = (task & 1023) << 4;                   // flattened b*T+t
  const int lane = tid & 31;
  const int col = lane & 15, half = lane >> 4;

  const __bf16* wsrc[3] = { wqt + (size_t)h * 64 * 512,
                            wkt + (size_t)h * 64 * 512,
                            wvt + (size_t)h * 64 * 512 };
  const int sn = tid >> 2, sck = tid & 3;              // staging coords

  v8f aq[4], ak[4], av[4];
#pragma unroll
  for (int n = 0; n < 4; ++n) { aq[n] = ZV8; ak[n] = ZV8; av[n] = ZV8; }

  for (int k0 = 0; k0 < 512; k0 += 32) {
    // stage 3 x (64 x 32) weight tiles: 768 16B chunks, 3 per thread
#pragma unroll
    for (int w = 0; w < 3; ++w)
      stage16(wsrc[w] + sn * 512 + k0 + sck * 8,
              wtile + w * WTILE + sn * WROW + sck * 8);
#if HAVE_ASYNC_LDS
    wait_async();
#endif
    __syncthreads();

    const v16bf a = load_a_frag(xbf + (size_t)m0 * 512 + k0, 512);
#pragma unroll
    for (int n = 0; n < 4; ++n) {
      // distinct B-frag locals: lets the scheduler batch the 6 ds_load_b128
      // and amortize s_wait_dscnt across the 3 WMMAs
      const v16bf bfq = load_b_frag(wtile + 0 * WTILE + (n * 16) * WROW, WROW);
      const v16bf bfk = load_b_frag(wtile + 1 * WTILE + (n * 16) * WROW, WROW);
      const v16bf bfv = load_b_frag(wtile + 2 * WTILE + (n * 16) * WROW, WROW);
      aq[n] = wmma_bf16(a, bfq, aq[n]);
      ak[n] = wmma_bf16(a, bfk, ak[n]);
      av[n] = wmma_bf16(a, bfv, av[n]);
    }
    __syncthreads();   // before next k-step overwrites the tiles
  }

  const int b = m0 >> 8, t0 = m0 & 255;
  const size_t bh = (size_t)(b * 8 + h);
  const float scale = 0.125f;                          // 1/sqrt(64)
#pragma unroll
  for (int n = 0; n < 4; ++n) {
    const int d = n * 16 + col;
#pragma unroll
    for (int r = 0; r < 8; ++r) {
      const int t = t0 + r + 8 * half;
      qbf[(bh * 256 + t) * 64 + d] = (__bf16)((aq[n][r] + bq[h * 64 + d]) * scale);
      kbf[(bh * 256 + t) * 64 + d] = (__bf16)(ak[n][r] + bk[h * 64 + d]);
      vtb[(bh * 64 + d) * 256 + t] = (__bf16)(av[n][r] + bv[h * 64 + d]);
    }
  }
}

// ---- Stage 2: flash attention --------------------------------------------
__global__ __launch_bounds__(256, 1)
void mha_attn(const __bf16* __restrict__ qbf, const __bf16* __restrict__ kbf,
              const __bf16* __restrict__ vtb, __bf16* __restrict__ comb) {
  __shared__ __bf16 plds[8][16 * 32];                  // per-wave P staging
  const int wave = threadIdx.x >> 5;
  const int lane = threadIdx.x & 31;
  const int task = blockIdx.x * 8 + wave;              // 8192 = 512 bh * 16
  const int bh = task >> 4;
  const int m0 = (task & 15) << 4;
  const int b = bh >> 3, h = bh & 7;
  const int col = lane & 15, half = lane >> 4;

  const __bf16* Q = qbf + (size_t)bh * 256 * 64;
  const __bf16* K = kbf + (size_t)bh * 256 * 64;
  const __bf16* V = vtb + (size_t)bh * 64 * 256;

  const v16bf aq0 = load_a_frag(Q + m0 * 64, 64);
  const v16bf aq1 = load_a_frag(Q + m0 * 64 + 32, 64);

  v8f o[4];
  float mr[8], lr[8];
#pragma unroll
  for (int n = 0; n < 4; ++n) o[n] = ZV8;
#pragma unroll
  for (int r = 0; r < 8; ++r) { mr[r] = -1e30f; lr[r] = 0.f; }

  __bf16* P = plds[wave];

  for (int s0 = 0; s0 < m0 + 16; s0 += 32) {           // causal block range
    // prefetch V fragments for this s-block: independent of the softmax
    // pipeline, so their global loads overlap QK^T WMMAs + exp VALU work
    v16bf bv[4];
#pragma unroll
    for (int n = 0; n < 4; ++n)
      bv[n] = load_b_frag(V + (n * 16) * 256 + s0, 256);

    v8f s[2];
#pragma unroll
    for (int t = 0; t < 2; ++t) {
      const v16bf bk0 = load_b_frag(K + (s0 + 16 * t) * 64, 64);
      const v16bf bk1 = load_b_frag(K + (s0 + 16 * t) * 64 + 32, 64);
      s[t] = ZV8;
      s[t] = wmma_bf16(aq0, bk0, s[t]);
      s[t] = wmma_bf16(aq1, bk1, s[t]);
    }
    float tm[8];
#pragma unroll
    for (int r = 0; r < 8; ++r) tm[r] = -1e30f;
#pragma unroll
    for (int t = 0; t < 2; ++t)
#pragma unroll
      for (int r = 0; r < 8; ++r) {
        const int cg = s0 + 16 * t + col;
        const int rg = m0 + r + 8 * half;
        const float v = (cg <= rg) ? s[t][r] : -1e30f;
        s[t][r] = v;
        tm[r] = fmaxf(tm[r], v);
      }
#pragma unroll
    for (int r = 0; r < 8; ++r) {
      tm[r] = xor_max16(tm[r]);
      const float mn = fmaxf(mr[r], tm[r]);
      const float alpha = __expf(mr[r] - mn);
      mr[r] = mn;
      lr[r] *= alpha;
#pragma unroll
      for (int n = 0; n < 4; ++n) o[n][r] *= alpha;
    }
    float rs[8];
#pragma unroll
    for (int r = 0; r < 8; ++r) rs[r] = 0.f;
#pragma unroll
    for (int t = 0; t < 2; ++t)
#pragma unroll
      for (int r = 0; r < 8; ++r) {
        const float p = __expf(s[t][r] - mr[r]);
        rs[r] += p;
        P[(r + 8 * half) * 32 + 16 * t + col] = (__bf16)p;
      }
#pragma unroll
    for (int r = 0; r < 8; ++r) lr[r] += xor_sum16(rs[r]);

    const v16bf ap = load_a_frag(P, 32);               // C->A relayout via LDS
#pragma unroll
    for (int n = 0; n < 4; ++n)
      o[n] = wmma_bf16(ap, bv[n], o[n]);
  }

#pragma unroll
  for (int n = 0; n < 4; ++n)
#pragma unroll
    for (int r = 0; r < 8; ++r) {
      const int t = m0 + r + 8 * half;
      const int c = h * 64 + n * 16 + col;
      comb[(size_t)(b * 256 + t) * 512 + c] = (__bf16)(o[n][r] / lr[r]);
    }
}

// ---- Stage 3: output projection ------------------------------------------
// Block = 8 waves sharing one 64-col weight strip (staged in LDS),
// 8 consecutive M-tiles.
__global__ __launch_bounds__(256, 1)
void mha_proj(const __bf16* __restrict__ comb, const __bf16* __restrict__ wot,
              const float* __restrict__ bo, float* __restrict__ out) {
  __shared__ __bf16 wtile[WTILE];                      // 5120 B
  const int tid = threadIdx.x;
  const int wave = tid >> 5;
  const int task = blockIdx.x * 8 + wave;              // 8192 = 8 strips * 1024
  const int n0 = (task >> 10) << 6;
  const int m0 = (task & 1023) << 4;
  const int lane = tid & 31;
  const int col = lane & 15, half = lane >> 4;
  const int sn = tid >> 2, sck = tid & 3;

  v8f acc[4];
#pragma unroll
  for (int n = 0; n < 4; ++n) acc[n] = ZV8;

  for (int k0 = 0; k0 < 512; k0 += 32) {
    stage16(wot + (size_t)(n0 + sn) * 512 + k0 + sck * 8,
            wtile + sn * WROW + sck * 8);
#if HAVE_ASYNC_LDS
    wait_async();
#endif
    __syncthreads();

    const v16bf a = load_a_frag(comb + (size_t)m0 * 512 + k0, 512);
    // load all 4 B-frags up front, then 4 back-to-back WMMAs
    v16bf bf[4];
#pragma unroll
    for (int n = 0; n < 4; ++n)
      bf[n] = load_b_frag(wtile + (n * 16) * WROW, WROW);
#pragma unroll
    for (int n = 0; n < 4; ++n)
      acc[n] = wmma_bf16(a, bf[n], acc[n]);
    __syncthreads();
  }

#pragma unroll
  for (int n = 0; n < 4; ++n) {
    const int c = n0 + n * 16 + col;
#pragma unroll
    for (int r = 0; r < 8; ++r)
      out[(size_t)(m0 + r + 8 * half) * 512 + c] = acc[n][r] + bo[c];
  }
}

// ---- launcher -------------------------------------------------------------
extern "C" void kernel_launch(void* const* d_in, const int* in_sizes, int n_in,
                              void* d_out, int out_size, void* d_ws, size_t ws_size,
                              hipStream_t stream) {
  const float* x  = (const float*)d_in[0];
  const float* wq = (const float*)d_in[1];
  const float* wk = (const float*)d_in[2];
  const float* wv = (const float*)d_in[3];
  const float* bq = (const float*)d_in[4];
  const float* bk = (const float*)d_in[5];
  const float* bv = (const float*)d_in[6];
  const float* wo = (const float*)d_in[7];
  const float* bo = (const float*)d_in[8];
  float* out = (float*)d_out;

  char* ws = (char*)d_ws;
  const size_t MB16 = 16u << 20;
  __bf16* xbf = (__bf16*)(ws);              // 16 MB; reused as 'combined'
  __bf16* qbf = (__bf16*)(ws + 1 * MB16);
  __bf16* kbf = (__bf16*)(ws + 2 * MB16);
  __bf16* vtb = (__bf16*)(ws + 3 * MB16);
  __bf16* wqt = (__bf16*)(ws + 4 * MB16);
  __bf16* wkt = (__bf16*)(ws + 4 * MB16 + 1 * 524288);
  __bf16* wvt = (__bf16*)(ws + 4 * MB16 + 2 * 524288);
  __bf16* wot = (__bf16*)(ws + 4 * MB16 + 3 * 524288);

  mha_cvt_x   <<<8192, 256, 0, stream>>>(x, xbf);
  mha_cvt_wqkv<<<1024, 256, 0, stream>>>(wq, wqt);
  mha_cvt_wqkv<<<1024, 256, 0, stream>>>(wk, wkt);
  mha_cvt_wqkv<<<1024, 256, 0, stream>>>(wv, wvt);
  mha_cvt_wo  <<<1024, 256, 0, stream>>>(wo, wot);

  mha_qkv <<<1024, 256, 0, stream>>>(xbf, wqt, wkt, wvt, bq, bk, bv, qbf, kbf, vtb);
  mha_attn<<<1024, 256, 0, stream>>>(qbf, kbf, vtb, xbf /* combined (alias) */);
  mha_proj<<<1024, 256, 0, stream>>>(xbf, wot, bo, out);
}
